// GSLModel_LabelDistr_58978490909109
// MI455X (gfx1250) — compile-verified, matrix operand order checked by reading, not run on hardware
//
#include <hip/hip_runtime.h>
#include <hip/hip_bf16.h>

// Problem constants (match reference).
#define N_NODES 8192
#define FEAT    512
#define C       40
#define CP      48          // classes padded to 3x16 WMMA tiles
#define HID     1024
#define HOPS    4
#define IN_DIM  672         // FEAT + HOPS*C ; 672 = 21*32
#define KSPLIT  8           // split-K for the hop GEMM (deterministic partials)

typedef __attribute__((ext_vector_type(16))) __bf16 v16bf;
typedef __attribute__((ext_vector_type(8)))  __bf16 v8bf;
typedef __attribute__((ext_vector_type(8)))  float  v8f;

union FragB16 { v16bf v; v8bf h[2]; };

__device__ __forceinline__ float sigmoidf_(float x) {
    return 1.0f / (1.0f + __expf(-x));
}

// ---------------------------------------------------------------------------
// Prep: A_bf16 = bf16(sigmoid(adj)), d_inv = 1/(rowsum + 1e-8). One block/row.
// ---------------------------------------------------------------------------
__global__ void k_prep_adj(const float* __restrict__ adj,
                           __bf16* __restrict__ Abf,
                           float* __restrict__ dinv) {
    __shared__ float red[256];
    const int row = blockIdx.x;
    const float* a = adj + (size_t)row * N_NODES;
    __bf16* o = Abf + (size_t)row * N_NODES;
    float s = 0.f;
    for (int j = threadIdx.x; j < N_NODES; j += 256) {
        float v = sigmoidf_(a[j]);
        o[j] = (__bf16)v;
        s += v;
    }
    red[threadIdx.x] = s;
    __syncthreads();
    for (int w = 128; w > 0; w >>= 1) {
        if (threadIdx.x < w) red[threadIdx.x] += red[threadIdx.x + w];
        __syncthreads();
    }
    if (threadIdx.x == 0) dinv[row] = 1.0f / (red[0] + 1e-8f);
}

// curT[c][n] = onehot[n][c] (bf16, transposed so B-fragments are contiguous-K)
__global__ void k_prep_curT(const float* __restrict__ onehot, __bf16* __restrict__ curT) {
    int idx = blockIdx.x * 256 + threadIdx.x;       // < CP*N_NODES
    int c = idx / N_NODES, n = idx % N_NODES;
    float v = (c < C) ? onehot[(size_t)n * C + c] : 0.f;
    curT[idx] = (__bf16)v;
}

// combined[:, 0:FEAT] = bf16(X)
__global__ void k_prep_X(const float* __restrict__ X, __bf16* __restrict__ comb) {
    int idx = blockIdx.x * 256 + threadIdx.x;       // < N_NODES*FEAT
    int i = idx / FEAT, f = idx % FEAT;
    comb[(size_t)i * IN_DIM + f] = (__bf16)X[idx];
}

// W1T[n][k] = bf16(W1[k][n])
__global__ void k_prep_W1T(const float* __restrict__ W1, __bf16* __restrict__ W1T) {
    int idx = blockIdx.x * 256 + threadIdx.x;       // < HID*IN_DIM
    int n = idx / IN_DIM, k = idx % IN_DIM;
    W1T[idx] = (__bf16)W1[(size_t)k * HID + n];
}

// W2T[c][k] = bf16(W2[k][c]), pad classes zeroed
__global__ void k_prep_W2T(const float* __restrict__ W2, __bf16* __restrict__ W2T) {
    int idx = blockIdx.x * 256 + threadIdx.x;       // < CP*HID
    int c = idx / HID, k = idx % HID;
    float v = (c < C) ? W2[(size_t)k * C + c] : 0.f;
    W2T[idx] = (__bf16)v;
}

// ---------------------------------------------------------------------------
// Hop GEMM: Yp[ks] += A[16-row tile, Kslice] @ curT^T  (bf16 WMMA, f32 acc)
// One wave = 16 rows x 48 cols, K slice of 1024. grid = (128, KSPLIT), 4 waves.
// ---------------------------------------------------------------------------
__global__ void k_hop_gemm(const __bf16* __restrict__ A,
                           const __bf16* __restrict__ Bt,
                           float* __restrict__ Yp) {
    const int lane = threadIdx.x & 31;
    const int wave = threadIdx.x >> 5;
    const int m0 = blockIdx.x * 64 + wave * 16;
    const int ks = blockIdx.y;
    const int k0 = ks * (N_NODES / KSPLIT);
    const int koff = (lane >> 4) * 8;               // ISA 16-bit A/B lane layout
    const int lm = lane & 15;

    const __bf16* arow = A  + (size_t)(m0 + lm) * N_NODES + k0 + koff;
    const __bf16* br0  = Bt + (size_t)(lm +  0) * N_NODES + k0 + koff;
    const __bf16* br1  = Bt + (size_t)(lm + 16) * N_NODES + k0 + koff;
    const __bf16* br2  = Bt + (size_t)(lm + 32) * N_NODES + k0 + koff;

    v8f acc0 = {}, acc1 = {}, acc2 = {};
    for (int k = 0; k < N_NODES / KSPLIT; k += 32) {
        FragB16 fa, f0, f1, f2;
        fa.h[0] = *(const v8bf*)(arow + k); fa.h[1] = *(const v8bf*)(arow + k + 16);
        f0.h[0] = *(const v8bf*)(br0 + k);  f0.h[1] = *(const v8bf*)(br0 + k + 16);
        f1.h[0] = *(const v8bf*)(br1 + k);  f1.h[1] = *(const v8bf*)(br1 + k + 16);
        f2.h[0] = *(const v8bf*)(br2 + k);  f2.h[1] = *(const v8bf*)(br2 + k + 16);
        acc0 = __builtin_amdgcn_wmma_f32_16x16x32_bf16(false, fa.v, false, f0.v, (short)0, acc0, false, false);
        acc1 = __builtin_amdgcn_wmma_f32_16x16x32_bf16(false, fa.v, false, f1.v, (short)0, acc1, false, false);
        acc2 = __builtin_amdgcn_wmma_f32_16x16x32_bf16(false, fa.v, false, f2.v, (short)0, acc2, false, false);
    }

    float* ybase = Yp + (size_t)ks * N_NODES * CP;
    const int rhalf = (lane >> 4) * 8;              // C/D layout: M = r + 8*(lane>=16)
    for (int r = 0; r < 8; ++r) {
        float* y = ybase + (size_t)(m0 + rhalf + r) * CP;
        y[ 0 + lm] = acc0[r];
        y[16 + lm] = acc1[r];
        y[32 + lm] = acc2[r];
    }
}

// ---------------------------------------------------------------------------
// Softmax epilogue: reduce split-K partials, scale by d_inv, softmax over 40,
// emit bf16 into curT (next hop's B) and into combined (MLP input).
// ---------------------------------------------------------------------------
__global__ void k_softmax(const float* __restrict__ Yp, const float* __restrict__ dinv,
                          __bf16* __restrict__ curT, __bf16* __restrict__ comb,
                          int col_off) {
    const int i = blockIdx.x * 256 + threadIdx.x;   // node id
    const float di = dinv[i];
    float v[C];
    float m = -1e30f;
    for (int c = 0; c < C; ++c) {
        float s = 0.f;
        for (int ks = 0; ks < KSPLIT; ++ks)
            s += Yp[((size_t)ks * N_NODES + i) * CP + c];
        s *= di;
        v[c] = s;
        m = fmaxf(m, s);
    }
    float sum = 0.f;
    for (int c = 0; c < C; ++c) { float e = __expf(v[c] - m); v[c] = e; sum += e; }
    const float inv = 1.0f / sum;
    for (int c = 0; c < C; ++c) {
        float p = v[c] * inv;
        curT[(size_t)c * N_NODES + i] = (__bf16)p;
        comb[(size_t)i * IN_DIM + col_off + c] = (__bf16)p;
    }
}

// ---------------------------------------------------------------------------
// MLP layer 1: H = relu(combined @ W1 + b1), bf16 in/out, f32 accumulate.
// One wave = 16 rows x 64 cols. grid = (128, 16), 4 waves/block.
// ---------------------------------------------------------------------------
__global__ void k_mlp1(const __bf16* __restrict__ A, const __bf16* __restrict__ Bt,
                       const float* __restrict__ bias, __bf16* __restrict__ H) {
    const int lane = threadIdx.x & 31;
    const int wave = threadIdx.x >> 5;
    const int m0 = blockIdx.x * 64 + wave * 16;
    const int n0 = blockIdx.y * 64;
    const int koff = (lane >> 4) * 8;
    const int lm = lane & 15;

    const __bf16* arow = A + (size_t)(m0 + lm) * IN_DIM + koff;
    const __bf16* br[4];
    for (int t = 0; t < 4; ++t)
        br[t] = Bt + (size_t)(n0 + t * 16 + lm) * IN_DIM + koff;

    v8f acc[4] = {};
    for (int k = 0; k < IN_DIM; k += 32) {
        FragB16 fa;
        fa.h[0] = *(const v8bf*)(arow + k); fa.h[1] = *(const v8bf*)(arow + k + 16);
        for (int t = 0; t < 4; ++t) {
            FragB16 fb;
            fb.h[0] = *(const v8bf*)(br[t] + k); fb.h[1] = *(const v8bf*)(br[t] + k + 16);
            acc[t] = __builtin_amdgcn_wmma_f32_16x16x32_bf16(false, fa.v, false, fb.v, (short)0, acc[t], false, false);
        }
    }

    const int rhalf = (lane >> 4) * 8;
    for (int t = 0; t < 4; ++t) {
        const int col = n0 + t * 16 + lm;
        const float bz = bias[col];
        for (int r = 0; r < 8; ++r) {
            float h = acc[t][r] + bz;
            h = h > 0.f ? h : 0.f;
            H[(size_t)(m0 + rhalf + r) * HID + col] = (__bf16)h;
        }
    }
}

// ---------------------------------------------------------------------------
// MLP layer 2: out = H @ W2 + b2 (f32 out, only 40 valid cols stored).
// One wave = 16 rows x 48 cols, K = 1024. grid = 128, 4 waves/block.
// ---------------------------------------------------------------------------
__global__ void k_mlp2(const __bf16* __restrict__ H, const __bf16* __restrict__ Bt,
                       const float* __restrict__ bias, float* __restrict__ out) {
    const int lane = threadIdx.x & 31;
    const int wave = threadIdx.x >> 5;
    const int m0 = blockIdx.x * 64 + wave * 16;
    const int koff = (lane >> 4) * 8;
    const int lm = lane & 15;

    const __bf16* arow = H  + (size_t)(m0 + lm) * HID + koff;
    const __bf16* br0  = Bt + (size_t)(lm +  0) * HID + koff;
    const __bf16* br1  = Bt + (size_t)(lm + 16) * HID + koff;
    const __bf16* br2  = Bt + (size_t)(lm + 32) * HID + koff;

    v8f a0 = {}, a1 = {}, a2 = {};
    for (int k = 0; k < HID; k += 32) {
        FragB16 fa, f0, f1, f2;
        fa.h[0] = *(const v8bf*)(arow + k); fa.h[1] = *(const v8bf*)(arow + k + 16);
        f0.h[0] = *(const v8bf*)(br0 + k);  f0.h[1] = *(const v8bf*)(br0 + k + 16);
        f1.h[0] = *(const v8bf*)(br1 + k);  f1.h[1] = *(const v8bf*)(br1 + k + 16);
        f2.h[0] = *(const v8bf*)(br2 + k);  f2.h[1] = *(const v8bf*)(br2 + k + 16);
        a0 = __builtin_amdgcn_wmma_f32_16x16x32_bf16(false, fa.v, false, f0.v, (short)0, a0, false, false);
        a1 = __builtin_amdgcn_wmma_f32_16x16x32_bf16(false, fa.v, false, f1.v, (short)0, a1, false, false);
        a2 = __builtin_amdgcn_wmma_f32_16x16x32_bf16(false, fa.v, false, f2.v, (short)0, a2, false, false);
    }

    const int rhalf = (lane >> 4) * 8;
    const float bz0 = bias[lm];
    const float bz1 = bias[16 + lm];
    const float bz2 = (lm < 8) ? bias[32 + lm] : 0.f;
    for (int r = 0; r < 8; ++r) {
        float* o = out + (size_t)(m0 + rhalf + r) * C;
        o[lm] = a0[r] + bz0;
        o[16 + lm] = a1[r] + bz1;
        if (lm < 8) o[32 + lm] = a2[r] + bz2;
    }
}

// ---------------------------------------------------------------------------
extern "C" void kernel_launch(void* const* d_in, const int* in_sizes, int n_in,
                              void* d_out, int out_size, void* d_ws, size_t ws_size,
                              hipStream_t stream) {
    (void)in_sizes; (void)n_in; (void)out_size; (void)ws_size;
    const float* X      = (const float*)d_in[0];
    const float* onehot = (const float*)d_in[1];
    const float* adj    = (const float*)d_in[2];
    const float* W1     = (const float*)d_in[3];
    const float* b1v    = (const float*)d_in[4];
    const float* W2     = (const float*)d_in[5];
    const float* b2v    = (const float*)d_in[6];
    float* out = (float*)d_out;
    char* ws = (char*)d_ws;

    constexpr size_t SZ_A    = (size_t)N_NODES * N_NODES * 2;       // 128 MB (fits in 192 MB L2)
    constexpr size_t SZ_CURT = (size_t)CP * N_NODES * 2;
    constexpr size_t SZ_Y    = (size_t)KSPLIT * N_NODES * CP * 4;
    constexpr size_t SZ_COMB = (size_t)N_NODES * IN_DIM * 2;
    constexpr size_t SZ_W1T  = (size_t)HID * IN_DIM * 2;
    constexpr size_t SZ_W2T  = (size_t)CP * HID * 2;
    constexpr size_t SZ_H    = (size_t)N_NODES * HID * 2;

    __bf16* Abf  = (__bf16*)(ws);
    __bf16* curT = (__bf16*)(ws + SZ_A);
    float*  Yp   = (float* )(ws + SZ_A + SZ_CURT);
    __bf16* comb = (__bf16*)(ws + SZ_A + SZ_CURT + SZ_Y);
    __bf16* W1T  = (__bf16*)(ws + SZ_A + SZ_CURT + SZ_Y + SZ_COMB);
    __bf16* W2T  = (__bf16*)(ws + SZ_A + SZ_CURT + SZ_Y + SZ_COMB + SZ_W1T);
    __bf16* H    = (__bf16*)(ws + SZ_A + SZ_CURT + SZ_Y + SZ_COMB + SZ_W1T + SZ_W2T);
    float*  dinv = (float* )(ws + SZ_A + SZ_CURT + SZ_Y + SZ_COMB + SZ_W1T + SZ_W2T + SZ_H);

    k_prep_adj<<<N_NODES, 256, 0, stream>>>(adj, Abf, dinv);
    k_prep_curT<<<(CP * N_NODES) / 256, 256, 0, stream>>>(onehot, curT);
    k_prep_X<<<(N_NODES * FEAT) / 256, 256, 0, stream>>>(X, comb);
    k_prep_W1T<<<(HID * IN_DIM) / 256, 256, 0, stream>>>(W1, W1T);
    k_prep_W2T<<<(CP * HID) / 256, 256, 0, stream>>>(W2, W2T);

    for (int hop = 0; hop < HOPS; ++hop) {
        k_hop_gemm<<<dim3(N_NODES / 64, KSPLIT), 128, 0, stream>>>(Abf, curT, Yp);
        k_softmax<<<N_NODES / 256, 256, 0, stream>>>(Yp, dinv, curT, comb, FEAT + hop * C);
    }

    k_mlp1<<<dim3(N_NODES / 64, HID / 64), 128, 0, stream>>>(comb, W1T, b1v, H);
    k_mlp2<<<N_NODES / 64, 128, 0, stream>>>(H, W2T, b2v, out);
}
